// LSTM_40638980554788
// MI455X (gfx1250) — matrix operand order repaired and checked by reading, not compiled, not run on hardware
//
#include <hip/hip_runtime.h>
#include <hip/hip_bf16.h>

#define H   2048
#define S   1024
#define FH  (4 * H)          // 8192
#define NBLK 256             // persistent blocks for recurrence
#define TPB  256

typedef __attribute__((ext_vector_type(16))) _Float16 v16h;
typedef __attribute__((ext_vector_type(8)))  float    v8f;

// ---------------- workspace layout (bytes) ----------------
static constexpr size_t OFF_WHH0 = 0;                       // 33554432  f16 [FH,H]
static constexpr size_t OFF_WHH1 = 33554432;                // 33554432  f16 [FH,H]
static constexpr size_t OFF_WIH1 = 67108864;                // 33554432  f16 [FH,H]
static constexpr size_t OFF_XP1  = 100663296;               // 33554432  f32 [S,FH]
static constexpr size_t OFF_Y0H  = 134217728;               // 4194304   f16 [S,H]
static constexpr size_t OFF_BS0  = 138412032;               // 32768     f32 [FH]
static constexpr size_t OFF_BS1  = 138444800;               // 32768     f32 [FH]
static constexpr size_t OFF_HB0  = 138477568;               // 16384     f32 [2,H]
static constexpr size_t OFF_HB1  = 138493952;               // 16384     f32 [2,H]
static constexpr size_t OFF_BAR  = 138510336;               // 256       u32 count,sense

// ---------------- grid-wide sense-reversing barrier ----------------
__device__ __forceinline__ void gridBarrier(unsigned* bar) {
    __syncthreads();
    if (threadIdx.x == 0) {
        unsigned* count = bar;
        unsigned* sense = bar + 1;
        __threadfence();
        unsigned s = __hip_atomic_load(sense, __ATOMIC_ACQUIRE, __HIP_MEMORY_SCOPE_AGENT);
        unsigned prev = __hip_atomic_fetch_add(count, 1u, __ATOMIC_ACQ_REL,
                                               __HIP_MEMORY_SCOPE_AGENT);
        if (prev == gridDim.x - 1) {
            __hip_atomic_store(count, 0u, __ATOMIC_RELAXED, __HIP_MEMORY_SCOPE_AGENT);
            __hip_atomic_fetch_add(sense, 1u, __ATOMIC_ACQ_REL, __HIP_MEMORY_SCOPE_AGENT);
        } else {
            while (__hip_atomic_load(sense, __ATOMIC_ACQUIRE,
                                     __HIP_MEMORY_SCOPE_AGENT) == s)
                __builtin_amdgcn_s_sleep(1);
        }
        __threadfence();
    }
    __syncthreads();
}

// ---------------- fp32 -> fp16 conversion ----------------
__global__ void k_cvt_f16(const float* __restrict__ src, _Float16* __restrict__ dst, int n) {
    int i = blockIdx.x * blockDim.x + threadIdx.x;
    int stride = gridDim.x * blockDim.x;
    for (; i < n; i += stride) dst[i] = (_Float16)src[i];
}

// ---------------- small prep: bias sums, h0 staging, barrier zero ----------------
__global__ void k_prep(const float* __restrict__ bih0, const float* __restrict__ bhh0,
                       const float* __restrict__ bih1, const float* __restrict__ bhh1,
                       const float* __restrict__ h0,   // [2*H] (batch dim squeezed)
                       float* __restrict__ bsum0, float* __restrict__ bsum1,
                       float* __restrict__ hbuf0, float* __restrict__ hbuf1,
                       unsigned* __restrict__ bar) {
    int i = blockIdx.x * blockDim.x + threadIdx.x;
    int stride = gridDim.x * blockDim.x;
    for (int r = i; r < FH; r += stride) {
        bsum0[r] = bih0[r] + bhh0[r];
        bsum1[r] = bih1[r] + bhh1[r];
    }
    for (int j = i; j < H; j += stride) {
        hbuf0[j] = h0[j];       // layer 0 initial h -> slot 0
        hbuf1[j] = h0[H + j];   // layer 1 initial h -> slot 0
    }
    if (i < 2) bar[i] = 0u;
}

// ---------------- persistent LSTM recurrence ----------------
// One wave owns one h-index j: computes all 4 gate dot-products (rows g*H+j of Whh,
// fp16 weights streamed from L2) against h (fp32, staged in LDS), then lane 0 does
// the gate nonlinearities and the c/h update. One grid barrier per timestep
// (h is ping-ponged between two slots, so read-slot and write-slot never alias).
__global__ __launch_bounds__(TPB)
void k_lstm_rec(const _Float16* __restrict__ Whh,   // [FH, H] f16
                const float* __restrict__ xp,       // [S, FH] (layer 1) or unused
                int use_xp,
                const float* __restrict__ x,        // [S, 2]   (layer 0)
                const float* __restrict__ Wih0,     // [FH, 2]  (layer 0)
                const float* __restrict__ bsum,     // [FH]     (layer 0)
                const float* __restrict__ c0,       // [H] initial c for this layer
                float* __restrict__ hbuf,           // [2, H] ping-pong
                _Float16* __restrict__ yout,        // [S, H] f16 (layer output)
                unsigned* __restrict__ bar) {
    __shared__ float hs[H];
    const int tid  = threadIdx.x;
    const int lane = tid & 31;
    const int w    = tid >> 5;
    const int j    = blockIdx.x * (H / NBLK) + w;   // H/NBLK == 8 == waves per block

    float c = c0[j];

    for (int t = 0; t < S; ++t) {
        // stage h_{t-1} into LDS
        const float* hsrc = hbuf + (size_t)(t & 1) * H;
        for (int i = tid; i < H; i += TPB) hs[i] = hsrc[i];
        __syncthreads();

        float z[4];
#pragma unroll
        for (int g = 0; g < 4; ++g) {
            const size_t row = (size_t)g * H + j;
            const uint4* wp = (const uint4*)(Whh + row * H);   // 8 f16 per uint4
            if (g < 3)   // pull next gate's row toward L2/L0 while we crunch this one
                __builtin_prefetch(Whh + (row + H) * H, 0, 1);
            float acc = 0.f;
#pragma unroll
            for (int i = 0; i < H / 256; ++i) {                // 8 iters: 32 lanes * 8 elems
                union { uint4 u; _Float16 h8[8]; } wv;
                wv.u = wp[i * 32 + lane];
                const float* hp = &hs[i * 256 + lane * 8];
                acc += (float)wv.h8[0] * hp[0];
                acc += (float)wv.h8[1] * hp[1];
                acc += (float)wv.h8[2] * hp[2];
                acc += (float)wv.h8[3] * hp[3];
                acc += (float)wv.h8[4] * hp[4];
                acc += (float)wv.h8[5] * hp[5];
                acc += (float)wv.h8[6] * hp[6];
                acc += (float)wv.h8[7] * hp[7];
            }
#pragma unroll
            for (int off = 16; off; off >>= 1) acc += __shfl_xor(acc, off, 32);
            z[g] = acc;
        }

        if (lane == 0) {
            float xv[4];
            if (use_xp) {
#pragma unroll
                for (int g = 0; g < 4; ++g) xv[g] = xp[(size_t)t * FH + (size_t)g * H + j];
            } else {
                float x0 = x[2 * t], x1 = x[2 * t + 1];
#pragma unroll
                for (int g = 0; g < 4; ++g) {
                    int row = g * H + j;
                    xv[g] = x0 * Wih0[2 * row] + x1 * Wih0[2 * row + 1] + bsum[row];
                }
            }
            float zi = z[0] + xv[0], zf = z[1] + xv[1];
            float zg = z[2] + xv[2], zo = z[3] + xv[3];
            float ig = 1.f / (1.f + __expf(-zi));
            float fg = 1.f / (1.f + __expf(-zf));
            float gg = tanhf(zg);
            float og = 1.f / (1.f + __expf(-zo));
            c = fg * c + ig * gg;
            float hnew = og * tanhf(c);
            hbuf[(size_t)((t & 1) ^ 1) * H + j] = hnew;
            yout[(size_t)t * H + j] = (_Float16)hnew;
        }
        gridBarrier(bar);
    }
}

// ---------------- WMMA GEMM: xp1[S,FH] = y0[S,H] @ Wih1[FH,H]^T + bsum1 ----------------
// A fragment (16-bit A 16x32): lanes 0-15 = rows M, elems 0-7 -> K 0..7, 8-15 -> K 16..23;
// lanes 16-31 same rows, K offset +8. B fragment (32x16): lane = column N,
// same K order as A: elems 0-7 -> K 8h..8h+7, 8-15 -> K 16+8h..16+8h+7 (Wih1 rows are
// contiguous in K, i.e. exactly B^T-friendly).
__global__ __launch_bounds__(TPB)
void k_gemm_xp1(const _Float16* __restrict__ A,   // [S, H]  f16 (y0)
                const _Float16* __restrict__ B,   // [FH, H] f16 (Wih1)
                const float* __restrict__ bias,   // [FH]
                float* __restrict__ Cout) {       // [S, FH] f32
    const int lane = threadIdx.x & 31;
    const int wv   = threadIdx.x >> 5;       // 8 waves: 2 (M) x 4 (N)
    const int l15  = lane & 15;
    const int hi   = lane >> 4;              // 0 or 1 (lane half)
    const int mblk = blockIdx.x & 31;        // 1024 / 32
    const int nblk = blockIdx.x >> 5;        // 8192 / 512
    const int m0   = mblk * 32 + (wv & 1) * 16;
    const int n0   = nblk * 512 + (wv >> 1) * 128;

    v8f acc[8];
#pragma unroll
    for (int j2 = 0; j2 < 8; ++j2) acc[j2] = v8f{0.f, 0.f, 0.f, 0.f, 0.f, 0.f, 0.f, 0.f};

    const _Float16* arow = A + (size_t)(m0 + l15) * H;

    for (int k0 = 0; k0 < H; k0 += 32) {
        union { uint4 u[2]; v16h v; } af;
        af.u[0] = *(const uint4*)(arow + k0 + hi * 8);        // K k0+h8 .. +7
        af.u[1] = *(const uint4*)(arow + k0 + 16 + hi * 8);   // K k0+16+h8 .. +7
#pragma unroll
        for (int j2 = 0; j2 < 8; ++j2) {
            const _Float16* brow = B + (size_t)(n0 + j2 * 16 + l15) * H + k0 + hi * 8;
            union { uint4 u[2]; v16h v; } bf;
            bf.u[0] = *(const uint4*)(brow);
            bf.u[1] = *(const uint4*)(brow + 16);
            acc[j2] = __builtin_amdgcn_wmma_f32_16x16x32_f16(
                false, af.v, false, bf.v, (short)0, acc[j2], false, false);
        }
    }

#pragma unroll
    for (int j2 = 0; j2 < 8; ++j2) {
        const int col = n0 + j2 * 16 + l15;
        const float bv = bias[col];
#pragma unroll
        for (int r = 0; r < 8; ++r) {
            const int m = m0 + hi * 8 + r;                    // C layout: VGPR r -> M
            Cout[(size_t)m * FH + col] = acc[j2][r] + bv;
        }
    }
}

// ---------------- final linear: out[2] = h1_final @ Wlin^T + blin ----------------
__global__ void k_final(const float* __restrict__ h1,    // [H] (hbuf1 slot 0)
                        const float* __restrict__ Wlin,  // [2, H]
                        const float* __restrict__ blin,  // [2]
                        float* __restrict__ out) {       // [2]
    __shared__ float red[2][8];
    const int tid = threadIdx.x, lane = tid & 31, w = tid >> 5;
    float p0 = 0.f, p1 = 0.f;
    for (int k = tid; k < H; k += 256) {
        float hv = h1[k];
        p0 += Wlin[k] * hv;
        p1 += Wlin[H + k] * hv;
    }
#pragma unroll
    for (int off = 16; off; off >>= 1) {
        p0 += __shfl_xor(p0, off, 32);
        p1 += __shfl_xor(p1, off, 32);
    }
    if (lane == 0) { red[0][w] = p0; red[1][w] = p1; }
    __syncthreads();
    if (tid < 2) {
        float s = 0.f;
        for (int i = 0; i < 8; ++i) s += red[tid][i];
        out[tid] = s + blin[tid];
    }
}

// ---------------- host launch ----------------
extern "C" void kernel_launch(void* const* d_in, const int* in_sizes, int n_in,
                              void* d_out, int out_size, void* d_ws, size_t ws_size,
                              hipStream_t stream) {
    const float* x    = (const float*)d_in[0];
    const float* h0   = (const float*)d_in[1];
    const float* c0   = (const float*)d_in[2];
    const float* Wih0 = (const float*)d_in[3];
    const float* Whh0 = (const float*)d_in[4];
    const float* bih0 = (const float*)d_in[5];
    const float* bhh0 = (const float*)d_in[6];
    const float* Wih1 = (const float*)d_in[7];
    const float* Whh1 = (const float*)d_in[8];
    const float* bih1 = (const float*)d_in[9];
    const float* bhh1 = (const float*)d_in[10];
    const float* Wlin = (const float*)d_in[11];
    const float* blin = (const float*)d_in[12];
    float* out = (float*)d_out;

    char* ws = (char*)d_ws;
    _Float16* whh0h = (_Float16*)(ws + OFF_WHH0);
    _Float16* whh1h = (_Float16*)(ws + OFF_WHH1);
    _Float16* wih1h = (_Float16*)(ws + OFF_WIH1);
    float*    xp1   = (float*)(ws + OFF_XP1);
    _Float16* y0h   = (_Float16*)(ws + OFF_Y0H);
    float*    bsum0 = (float*)(ws + OFF_BS0);
    float*    bsum1 = (float*)(ws + OFF_BS1);
    float*    hbuf0 = (float*)(ws + OFF_HB0);
    float*    hbuf1 = (float*)(ws + OFF_HB1);
    unsigned* bar   = (unsigned*)(ws + OFF_BAR);

    // 1. weights -> fp16 (all three big matrices are L2-resident in fp16)
    k_cvt_f16<<<2048, 256, 0, stream>>>(Whh0, whh0h, FH * H);
    k_cvt_f16<<<2048, 256, 0, stream>>>(Whh1, whh1h, FH * H);
    k_cvt_f16<<<2048, 256, 0, stream>>>(Wih1, wih1h, FH * H);

    // 2. bias sums, initial h staging, barrier reset (every call -> deterministic)
    k_prep<<<64, 256, 0, stream>>>(bih0, bhh0, bih1, bhh1, h0,
                                   bsum0, bsum1, hbuf0, hbuf1, bar);

    // 3. layer-0 recurrence (input projection folded in on the fly, F=2)
    k_lstm_rec<<<NBLK, TPB, 0, stream>>>(whh0h, nullptr, 0, x, Wih0, bsum0,
                                         c0 /* layer 0 */, hbuf0, y0h, bar);

    // 4. layer-1 input projection: one big fp16 WMMA GEMM (bias folded)
    k_gemm_xp1<<<512, TPB, 0, stream>>>(y0h, wih1h, bsum1, xp1);

    // 5. layer-1 recurrence (y history dumped back into y0h; only final h matters)
    k_lstm_rec<<<NBLK, TPB, 0, stream>>>(whh1h, xp1, 1, x, Wih0, bsum1,
                                         c0 + H /* layer 1 */, hbuf1, y0h, bar);

    // 6. final linear; S even -> final h sits in hbuf1 slot 0
    k_final<<<1, 256, 0, stream>>>(hbuf1, Wlin, blin, out);
}
